// Intra_ResNet_84524956385356
// MI455X (gfx1250) — matrix operand order repaired and checked by reading, not compiled, not run on hardware
//
#include <hip/hip_runtime.h>

// ---------------- constants ----------------
static constexpr int LDIM = 384;
static constexpr int NPIX = LDIM * LDIM;   // 147456
static constexpr int CH   = 64;
static constexpr int D1   = 788;
static constexpr int D2   = 210;

// ---------------- types / WMMA helpers ----------------
typedef __attribute__((ext_vector_type(16))) __bf16 v16bf;
typedef __attribute__((ext_vector_type(8)))  __bf16 v8bf;
typedef __attribute__((ext_vector_type(8)))  float  v8f;

__device__ __forceinline__ float lrelu(float x) { return x > 0.f ? x : 0.01f * x; }

__device__ __forceinline__ v8f wmma_bf16(v16bf a, v16bf b, v8f c) {
  // D = A(16x32) * B(32x16) + C, f32 accum
  return __builtin_amdgcn_wmma_f32_16x16x32_bf16(false, a, false, b, (short)0, c, false, false);
}

// A fragment (16x32 bf16, row-major [M][K] in LDS): lane l holds row m=l%16,
// K-span [kb + hi*16 .. +15]  (hi = l>=16)
__device__ __forceinline__ v16bf load_a_frag(const __bf16* base, int row_stride,
                                             int kbase, int lane) {
  const int m  = lane & 15;
  const int hi = (lane >> 4) & 1;
  const __bf16* rp = base + m * row_stride + kbase + hi * 16;
  union { v16bf v; v8bf h[2]; } u;
  u.h[0] = *(const v8bf*)(rp);
  u.h[1] = *(const v8bf*)(rp + 8);
  return u.v;
}

// ---------------- K1: instance-norm of x_1d over L ----------------
__global__ void k_inorm1d(const float* __restrict__ x, float* __restrict__ y) {
  const int c = blockIdx.x, tid = threadIdx.x;          // 788 blocks x 128
  __shared__ float ps[128], ps2[128];
  const float* xp = x + (size_t)c * LDIM;
  float s = 0.f, s2 = 0.f;
  for (int j = tid; j < LDIM; j += 128) { float v = xp[j]; s += v; s2 += v * v; }
  ps[tid] = s; ps2[tid] = s2; __syncthreads();
  for (int st = 64; st > 0; st >>= 1) {
    if (tid < st) { ps[tid] += ps[tid + st]; ps2[tid] += ps2[tid + st]; }
    __syncthreads();
  }
  float mu = ps[0] / LDIM;
  float rs = rsqrtf(ps2[0] / LDIM - mu * mu + 1e-5f);
  float* yp = y + (size_t)c * LDIM;
  for (int j = tid; j < LDIM; j += 128) yp[j] = (xp[j] - mu) * rs;
}

// ---- K2: row/col = W1a/W1b @ x1n, plus closed-form pair1 norm params ----
__global__ void k_rowcol(const float* __restrict__ x1n, const float* __restrict__ W1,
                         const float* __restrict__ g1,  const float* __restrict__ b1,
                         float* __restrict__ ra, float* __restrict__ ca,
                         float* __restrict__ off1) {
  const int c = blockIdx.x, tid = threadIdx.x;          // 64 blocks x 128
  __shared__ float row[LDIM], col[LDIM];
  __shared__ float pr[128], pr2[128], pc[128], pc2[128];
  const float* wa = W1 + (size_t)c * (2 * D1);
  const float* wb = wa + D1;
  for (int j = tid; j < LDIM; j += 128) {
    float sr = 0.f, sc = 0.f;
    for (int d = 0; d < D1; ++d) {
      float xv = x1n[(size_t)d * LDIM + j];
      sr += wa[d] * xv; sc += wb[d] * xv;
    }
    row[j] = sr; col[j] = sc;
  }
  __syncthreads();
  float sr = 0, sr2 = 0, sc = 0, sc2 = 0;
  for (int j = tid; j < LDIM; j += 128) {
    float r = row[j], q = col[j];
    sr += r; sr2 += r * r; sc += q; sc2 += q * q;
  }
  pr[tid] = sr; pr2[tid] = sr2; pc[tid] = sc; pc2[tid] = sc2; __syncthreads();
  for (int st = 64; st > 0; st >>= 1) {
    if (tid < st) {
      pr[tid] += pr[tid + st]; pr2[tid] += pr2[tid + st];
      pc[tid] += pc[tid + st]; pc2[tid] += pc2[tid + st];
    }
    __syncthreads();
  }
  float mr = pr[0] / LDIM, mc = pc[0] / LDIM;
  float vr = pr2[0] / LDIM - mr * mr, vc = pc2[0] / LDIM - mc * mc;
  float rs = rsqrtf(vr + vc + 1e-5f);            // Var(row_i+col_j) = Var_r + Var_c
  float scale = g1[c] * rs;
  for (int j = tid; j < LDIM; j += 128) {
    ra[(size_t)c * LDIM + j] = row[j] * scale;
    ca[(size_t)c * LDIM + j] = col[j] * scale;
  }
  if (tid == 0) off1[c] = b1[c] - (mr + mc) * scale;
}

// ---------------- K3: per-channel stats of x_2d ----------------
__global__ void k_stats2d(const float* __restrict__ x2, float* __restrict__ mu2,
                          float* __restrict__ rs2) {
  const int d = blockIdx.x, tid = threadIdx.x;          // 210 blocks x 256
  __shared__ float ps[256], ps2[256];
  const float* xp = x2 + (size_t)d * NPIX;
  float s = 0.f, s2 = 0.f;
  for (int i = tid; i < NPIX / 4; i += 256) {           // 144 exact iters, float4
    float4 v = *(const float4*)(xp + 4 * i);
    s  += v.x + v.y + v.z + v.w;
    s2 += v.x * v.x + v.y * v.y + v.z * v.z + v.w * v.w;
  }
  ps[tid] = s; ps2[tid] = s2; __syncthreads();
  for (int st = 128; st > 0; st >>= 1) {
    if (tid < st) { ps[tid] += ps[tid + st]; ps2[tid] += ps2[tid + st]; }
    __syncthreads();
  }
  if (tid == 0) {
    float mu = ps[0] / NPIX;
    mu2[d] = mu;
    rs2[d] = rsqrtf(ps2[0] / NPIX - mu * mu + 1e-5f);
  }
}

// -------- one-time weight prep: W3 -> bf16; res_w -> bf16 [conv][tap][c][d] --------
__global__ void k_prep_w3(const float* __restrict__ W3, __bf16* __restrict__ w3b) {
  int idx = blockIdx.x * 256 + threadIdx.x;             // 8192
  w3b[idx] = (__bf16)W3[idx];
}
__global__ void k_prep_convw(const float* __restrict__ rw, __bf16* __restrict__ cw) {
  int idx = blockIdx.x * 256 + threadIdx.x;             // 368640 total (1440 blocks)
  if (idx >= 10 * 9 * 64 * 64) return;
  int d  = idx & 63;
  int c  = (idx >> 6) & 63;
  int t  = (idx >> 12) % 9;
  int lk = idx / (9 * 4096);
  cw[idx] = (__bf16)rw[(size_t)((lk * 64 + c) * 64 + d) * 9 + t];
}

// ---- runtime weight prep: W2eff = W2*rs2 (bf16, K padded to 224) + folded bias ----
__global__ void k_prep2(const float* __restrict__ W2, const float* __restrict__ mu2,
                        const float* __restrict__ rs2, __bf16* __restrict__ w2b,
                        float* __restrict__ bias2) {
  const int tid = threadIdx.x;                          // 1 block x 256
  for (int idx = tid; idx < 64 * 224; idx += 256) {
    int c = idx / 224, d = idx % 224;
    float v = (d < D2) ? W2[(size_t)c * D2 + d] * rs2[d] : 0.f;
    w2b[idx] = (__bf16)v;
  }
  if (tid < 64) {
    float s = 0.f;
    for (int d = 0; d < D2; ++d) s += W2[(size_t)tid * D2 + d] * rs2[d] * mu2[d];
    bias2[tid] = -s;
  }
}

// -------- K5: pair2 = (W2*rs2) @ x2 + bias2  [WMMA bf16, double-buffered B] --------
__device__ __forceinline__ void stage_x2(const float* __restrict__ x2, int kc,
                                         int pixbase, int tid, __bf16* __restrict__ dst) {
  float4 v[4];
#pragma unroll
  for (int it = 0; it < 4; ++it) {                      // batch all 4 loads first
    int idx = tid + it * 256;
    int dl = idx >> 5, q = idx & 31;
    int d = kc * 32 + dl;
    v[it] = make_float4(0.f, 0.f, 0.f, 0.f);
    if (d < D2) v[it] = *(const float4*)(x2 + (size_t)d * NPIX + pixbase + 4 * q);
  }
#pragma unroll
  for (int it = 0; it < 4; ++it) {                      // then convert + store
    int idx = tid + it * 256;
    int dl = idx >> 5, q = idx & 31;
    __bf16* p = dst + (4 * q) * 32 + dl;
    p[0]  = (__bf16)v[it].x; p[32] = (__bf16)v[it].y;
    p[64] = (__bf16)v[it].z; p[96] = (__bf16)v[it].w;
  }
}

__global__ void __launch_bounds__(256) k_gemm_pair2(
    const float* __restrict__ x2, const __bf16* __restrict__ w2b,
    const float* __restrict__ bias2, float* __restrict__ P2,
    float* __restrict__ gsum, float* __restrict__ gssq) {
  __shared__ __align__(16) __bf16 wA[64 * 224];         // [c][k]   28 KB
  __shared__ __align__(16) __bf16 xB[2][128 * 32];      // [p][k]   2 x 8 KB
  __shared__ float ssum[64], sssq[64];
  const int tid = threadIdx.x;
  const int pixbase = blockIdx.x * 128;
  if (tid < 64) { ssum[tid] = 0.f; sssq[tid] = 0.f; }
  {
    const uint4* src = (const uint4*)w2b;               // 1792 x 16B
    uint4* dst = (uint4*)wA;
#pragma unroll
    for (int it = 0; it < 7; ++it) dst[tid + it * 256] = src[tid + it * 256];
  }
  stage_x2(x2, 0, pixbase, tid, xB[0]);
  const int lane = tid & 31, w = tid >> 5;
  const int cbase = (w & 3) * 16, pbw = (w >> 2) * 64;
  const int n = lane & 15, hi = (lane >> 4) & 1;
  v8f zero = {0.f, 0.f, 0.f, 0.f, 0.f, 0.f, 0.f, 0.f};
  v8f acc[4] = {zero, zero, zero, zero};
  __syncthreads();
  for (int kc = 0; kc < 7; ++kc) {
    if (kc < 6) stage_x2(x2, kc + 1, pixbase, tid, xB[(kc + 1) & 1]);
    const __bf16* cur = xB[kc & 1];
    v16bf a = load_a_frag(wA + cbase * 224, 224, kc * 32, lane);
#pragma unroll
    for (int pt = 0; pt < 4; ++pt) {
      v16bf b = *(const v16bf*)&cur[(pbw + pt * 16 + n) * 32 + hi * 16];
      acc[pt] = wmma_bf16(a, b, acc[pt]);
    }
    __syncthreads();
  }
  union { v8f v; float f[8]; } u[4];
#pragma unroll
  for (int pt = 0; pt < 4; ++pt) u[pt].v = acc[pt];
#pragma unroll
  for (int r = 0; r < 8; ++r) {
    int c = cbase + hi * 8 + r;
    float b = bias2[c], s = 0.f, q = 0.f;
#pragma unroll
    for (int pt = 0; pt < 4; ++pt) {
      float val = u[pt].f[r] + b;
      P2[(size_t)c * NPIX + pixbase + pbw + pt * 16 + n] = val;
      s += val; q += val * val;
    }
    atomicAdd(&ssum[c], s); atomicAdd(&sssq[c], q);
  }
  __syncthreads();
  if (tid < 64) { atomicAdd(&gsum[tid], ssum[tid]); atomicAdd(&gssq[tid], sssq[tid]); }
}

// ---- K7: pair = W3 @ [lrelu(norm(pair1)); lrelu(norm(pair2))]  (K=128, WMMA) ----
__device__ __forceinline__ void stage_pair_b(
    const float* __restrict__ ra, const float* __restrict__ ca,
    const float* __restrict__ off1, const float* __restrict__ P2,
    const float* __restrict__ s2, const float* __restrict__ t2,
    int kc, int i, int j0, int pixbase, int tid, __bf16* __restrict__ dst) {
  float4 r[4]; float sa[4], sb[4];
#pragma unroll
  for (int it = 0; it < 4; ++it) {                      // batch loads; v = r*sb + sa
    int idx = tid + it * 256;
    int dl = idx >> 5, q = idx & 31;
    if (kc < 2) {                                       // pair1 channels, closed form
      int e = kc * 32 + dl;
      r[it]  = *(const float4*)(ca + e * LDIM + j0 + 4 * q);
      sa[it] = ra[e * LDIM + i] + off1[e];
      sb[it] = 1.f;
    } else {                                            // normalized pair2 channels
      int e = (kc - 2) * 32 + dl;
      r[it]  = *(const float4*)(P2 + (size_t)e * NPIX + pixbase + 4 * q);
      sa[it] = t2[e];
      sb[it] = s2[e];
    }
  }
#pragma unroll
  for (int it = 0; it < 4; ++it) {
    int idx = tid + it * 256;
    int dl = idx >> 5, q = idx & 31;
    __bf16* p = dst + (4 * q) * 32 + dl;
    p[0]  = (__bf16)lrelu(r[it].x * sb[it] + sa[it]);
    p[32] = (__bf16)lrelu(r[it].y * sb[it] + sa[it]);
    p[64] = (__bf16)lrelu(r[it].z * sb[it] + sa[it]);
    p[96] = (__bf16)lrelu(r[it].w * sb[it] + sa[it]);
  }
}

__global__ void __launch_bounds__(256) k_gemm_pair(
    const __bf16* __restrict__ w3b, const float* __restrict__ ra,
    const float* __restrict__ ca, const float* __restrict__ off1,
    const float* __restrict__ P2, const float* __restrict__ s2,
    const float* __restrict__ t2, float* __restrict__ PAIR,
    float* __restrict__ gsum, float* __restrict__ gssq) {
  __shared__ __align__(16) __bf16 wA[64 * 128];         // 16 KB
  __shared__ __align__(16) __bf16 xB[2][128 * 32];      // 2 x 8 KB
  __shared__ float ssum[64], sssq[64];
  const int tid = threadIdx.x;
  const int pixbase = blockIdx.x * 128;
  const int i = pixbase / LDIM, j0 = pixbase % LDIM;    // 128 | 384 -> const i
  if (tid < 64) { ssum[tid] = 0.f; sssq[tid] = 0.f; }
  {
    const uint4* src = (const uint4*)w3b;               // 1024 x 16B
    uint4* dst = (uint4*)wA;
#pragma unroll
    for (int it = 0; it < 4; ++it) dst[tid + it * 256] = src[tid + it * 256];
  }
  stage_pair_b(ra, ca, off1, P2, s2, t2, 0, i, j0, pixbase, tid, xB[0]);
  const int lane = tid & 31, w = tid >> 5;
  const int cbase = (w & 3) * 16, pbw = (w >> 2) * 64;
  const int n = lane & 15, hi = (lane >> 4) & 1;
  v8f zero = {0.f, 0.f, 0.f, 0.f, 0.f, 0.f, 0.f, 0.f};
  v8f acc[4] = {zero, zero, zero, zero};
  __syncthreads();
  for (int kc = 0; kc < 4; ++kc) {
    if (kc < 3) stage_pair_b(ra, ca, off1, P2, s2, t2, kc + 1, i, j0, pixbase, tid,
                             xB[(kc + 1) & 1]);
    const __bf16* cur = xB[kc & 1];
    v16bf a = load_a_frag(wA + cbase * 128, 128, kc * 32, lane);
#pragma unroll
    for (int pt = 0; pt < 4; ++pt) {
      v16bf b = *(const v16bf*)&cur[(pbw + pt * 16 + n) * 32 + hi * 16];
      acc[pt] = wmma_bf16(a, b, acc[pt]);
    }
    __syncthreads();
  }
  union { v8f v; float f[8]; } u[4];
#pragma unroll
  for (int pt = 0; pt < 4; ++pt) u[pt].v = acc[pt];
#pragma unroll
  for (int r = 0; r < 8; ++r) {
    int c = cbase + hi * 8 + r;
    float s = 0.f, q = 0.f;
#pragma unroll
    for (int pt = 0; pt < 4; ++pt) {
      float val = u[pt].f[r];
      PAIR[(size_t)c * NPIX + pixbase + pbw + pt * 16 + n] = val;
      s += val; q += val * val;
    }
    atomicAdd(&ssum[c], s); atomicAdd(&sssq[c], q);
  }
  __syncthreads();
  if (tid < 64) { atomicAdd(&gsum[tid], ssum[tid]); atomicAdd(&gssq[tid], sssq[tid]); }
}

// -------- conv3x3 (dilated) as 9-tap implicit GEMM, 64ch x 128-pixel tiles --------
__global__ void __launch_bounds__(256) k_conv3x3(
    const float* __restrict__ Xin, const __bf16* __restrict__ Wt,  // [9][64][64] bf16
    const float* __restrict__ bc, int dil, float* __restrict__ Yraw,
    float* __restrict__ gsum, float* __restrict__ gssq) {
  __shared__ __align__(16) __bf16 lw[64 * 64];          // per-tap [c][d]  8 KB
  __shared__ __align__(16) __bf16 lx[3 * 136 * 64];     // [ry][col][d]   52 KB
  __shared__ float ssum[64], sssq[64];
  const int tid = threadIdx.x;
  const int bi  = blockIdx.x;                           // 384 rows * 3 col-tiles
  const int i   = bi / 3;
  const int j0  = (bi % 3) * 128;
  if (tid < 64) { ssum[tid] = 0.f; sssq[tid] = 0.f; }
  // stage input halo rows, fixed window gj = j0-4+cx (16B-aligned float4 quads),
  // batched 4 loads at a time (6528 quads total, tail predicated)
  for (int base = 0; base < 6528; base += 1024) {
    float4 v[4];
#pragma unroll
    for (int k = 0; k < 4; ++k) {
      int idx = base + k * 256 + tid;
      v[k] = make_float4(0.f, 0.f, 0.f, 0.f);
      if (idx < 6528) {
        int d   = idx / 102;                            // 102 = 3*34 quads per channel
        int rem = idx % 102;
        int ry  = rem / 34, qx = rem % 34;
        int gi = i + (ry - 1) * dil;
        int gj = j0 - 4 + 4 * qx;
        if (gi >= 0 && gi < LDIM && gj >= 0 && gj < LDIM)
          v[k] = *(const float4*)(Xin + (size_t)d * NPIX + gi * LDIM + gj);
      }
    }
#pragma unroll
    for (int k = 0; k < 4; ++k) {
      int idx = base + k * 256 + tid;
      if (idx < 6528) {
        int d   = idx / 102;
        int rem = idx % 102;
        int ry  = rem / 34, qx = rem % 34;
        __bf16* p = &lx[(ry * 136 + 4 * qx) * 64 + d];
        p[0]   = (__bf16)v[k].x; p[64]  = (__bf16)v[k].y;
        p[128] = (__bf16)v[k].z; p[192] = (__bf16)v[k].w;
      }
    }
  }
  const int lane = tid & 31, w = tid >> 5;
  const int cbase = (w & 3) * 16, pbw = (w >> 2) * 64;
  const int n = lane & 15, hi = (lane >> 4) & 1;
  v8f zero = {0.f, 0.f, 0.f, 0.f, 0.f, 0.f, 0.f, 0.f};
  v8f acc[4] = {zero, zero, zero, zero};
  for (int t = 0; t < 9; ++t) {
    __syncthreads();                                    // lw reads of tap t-1 done
    {                                                   // 8 KB b128 copy of tap t
      const uint4* src = (const uint4*)(Wt + t * 4096);
      uint4* dst = (uint4*)lw;
      uint4 w0 = src[tid], w1 = src[tid + 256];
      dst[tid] = w0; dst[tid + 256] = w1;
    }
    __syncthreads();
    const int ry  = t / 3;
    const int dxo = 4 + (t % 3 - 1) * dil;              // lx col = p + dxo in [0,135]
#pragma unroll
    for (int kc = 0; kc < 2; ++kc) {
      v16bf a = load_a_frag(lw + cbase * 64, 64, kc * 32, lane);
#pragma unroll
      for (int pt = 0; pt < 4; ++pt) {
        int col = pbw + pt * 16 + n + dxo;
        v16bf b = *(const v16bf*)&lx[(ry * 136 + col) * 64 + kc * 32 + hi * 16];
        acc[pt] = wmma_bf16(a, b, acc[pt]);
      }
    }
  }
  __syncthreads();
  union { v8f v; float f[8]; } u[4];
#pragma unroll
  for (int pt = 0; pt < 4; ++pt) u[pt].v = acc[pt];
#pragma unroll
  for (int r = 0; r < 8; ++r) {
    int c = cbase + hi * 8 + r;
    float b = bc[c], s = 0.f, q = 0.f;
#pragma unroll
    for (int pt = 0; pt < 4; ++pt) {
      float val = u[pt].f[r] + b;
      Yraw[(size_t)c * NPIX + i * LDIM + j0 + pbw + pt * 16 + n] = val;
      s += val; q += val * val;
    }
    atomicAdd(&ssum[c], s); atomicAdd(&sssq[c], q);
  }
  __syncthreads();
  if (tid < 64) { atomicAdd(&gsum[tid], ssum[tid]); atomicAdd(&gssq[tid], sssq[tid]); }
}

// ---------------- finalize stats -> scale/shift ----------------
__global__ void k_finalize(const float* __restrict__ sum, const float* __restrict__ ssq,
                           const float* __restrict__ gamma, const float* __restrict__ beta,
                           float* __restrict__ s, float* __restrict__ t) {
  int c = threadIdx.x;                                  // 64 threads
  float mu  = sum[c] / (float)NPIX;
  float var = ssq[c] / (float)NPIX - mu * mu;
  float rs  = rsqrtf(var + 1e-5f);
  float sc  = gamma[c] * rs;
  s[c] = sc; t[c] = beta[c] - mu * sc;
}

// ---------------- elementwise apply (optionally + residual), float4 ----------------
__global__ void k_apply(const float* __restrict__ raw, const float* __restrict__ s,
                        const float* __restrict__ t, float* __restrict__ dst) {
  int c = blockIdx.y;
  size_t idx = (size_t)c * NPIX + (blockIdx.x * 256 + threadIdx.x) * 4;
  float sc = s[c], tc = t[c];
  float4 v = *(const float4*)(raw + idx);
  v.x = lrelu(v.x * sc + tc); v.y = lrelu(v.y * sc + tc);
  v.z = lrelu(v.z * sc + tc); v.w = lrelu(v.w * sc + tc);
  *(float4*)(dst + idx) = v;
}
__global__ void k_apply_res(const float* __restrict__ raw, const float* __restrict__ s,
                            const float* __restrict__ t, const float* __restrict__ res,
                            float* __restrict__ dst) {
  int c = blockIdx.y;
  size_t idx = (size_t)c * NPIX + (blockIdx.x * 256 + threadIdx.x) * 4;
  float sc = s[c], tc = t[c];
  float4 v = *(const float4*)(raw + idx);
  float4 r = *(const float4*)(res + idx);
  v.x = lrelu(v.x * sc + tc) + r.x; v.y = lrelu(v.y * sc + tc) + r.y;
  v.z = lrelu(v.z * sc + tc) + r.z; v.w = lrelu(v.w * sc + tc) + r.w;
  *(float4*)(dst + idx) = v;
}

// ---------------- host orchestration ----------------
extern "C" void kernel_launch(void* const* d_in, const int* in_sizes, int n_in,
                              void* d_out, int out_size, void* d_ws, size_t ws_size,
                              hipStream_t stream) {
  const float* x1d = (const float*)d_in[0];
  const float* x2d = (const float*)d_in[1];
  const float* W1  = (const float*)d_in[2];
  const float* g1  = (const float*)d_in[3];
  const float* b1  = (const float*)d_in[4];
  const float* W2  = (const float*)d_in[5];
  const float* g2  = (const float*)d_in[6];
  const float* b2  = (const float*)d_in[7];
  const float* W3  = (const float*)d_in[8];
  const float* g3  = (const float*)d_in[9];
  const float* b3  = (const float*)d_in[10];
  const float* rw  = (const float*)d_in[11];  // (5,2,64,64,3,3)
  const float* rb  = (const float*)d_in[12];  // (5,2,64)
  const float* rg  = (const float*)d_in[13];
  const float* rbe = (const float*)d_in[14];
  float* out = (float*)d_out;

  char* ws = (char*)d_ws;
  size_t off = 0;
  auto allocB = [&](size_t bytes) -> char* {
    char* p = ws + off;
    off = (off + bytes + 255) & ~(size_t)255;
    return p;
  };
  auto allocF = [&](size_t n) -> float* { return (float*)allocB(n * sizeof(float)); };

  float*  x1n   = allocF((size_t)D1 * LDIM);
  float*  ra    = allocF((size_t)CH * LDIM);
  float*  ca    = allocF((size_t)CH * LDIM);
  float*  off1  = allocF(64);
  float*  mu2   = allocF(256);
  float*  rs2   = allocF(256);
  float*  bias2 = allocF(64);
  float*  s2 = allocF(64); float* t2 = allocF(64);
  float*  s3 = allocF(64); float* t3 = allocF(64);
  float*  sc = allocF(64); float* tc = allocF(64);
  float*  stats = allocF(512);
  float*  gsum = stats;        // [256]
  float*  gssq = stats + 256;  // [256]
  __bf16* w2b  = (__bf16*)allocB((size_t)64 * 224 * 2);
  __bf16* w3b  = (__bf16*)allocB((size_t)64 * 128 * 2);
  __bf16* cw   = (__bf16*)allocB((size_t)10 * 9 * 64 * 64 * 2);
  float*  BUFA = allocF((size_t)CH * NPIX);   // raw GEMM/conv outputs
  float*  BUFB = allocF((size_t)CH * NPIX);   // pair / mid activations
  float*  BUFC = allocF((size_t)CH * NPIX);   // layer input / residual

  const dim3 agrid(NPIX / 1024, CH);          // float4 apply
  const int DIL[5] = {1, 2, 4, 2, 1};

  k_prep_w3  <<<32, 256, 0, stream>>>(W3, w3b);
  k_prep_convw<<<1440, 256, 0, stream>>>(rw, cw);
  k_inorm1d  <<<D1, 128, 0, stream>>>(x1d, x1n);
  k_rowcol   <<<CH, 128, 0, stream>>>(x1n, W1, g1, b1, ra, ca, off1);
  k_stats2d  <<<D2, 256, 0, stream>>>(x2d, mu2, rs2);
  k_prep2    <<<1, 256, 0, stream>>>(W2, mu2, rs2, w2b, bias2);

  hipMemsetAsync(stats, 0, 512 * sizeof(float), stream);
  k_gemm_pair2<<<NPIX / 128, 256, 0, stream>>>(x2d, w2b, bias2, BUFA, gsum, gssq);
  k_finalize<<<1, 64, 0, stream>>>(gsum, gssq, g2, b2, s2, t2);

  hipMemsetAsync(stats, 0, 512 * sizeof(float), stream);
  k_gemm_pair<<<NPIX / 128, 256, 0, stream>>>(w3b, ra, ca, off1, BUFA, s2, t2, BUFB,
                                              gsum, gssq);
  k_finalize<<<1, 64, 0, stream>>>(gsum, gssq, g3, b3, s3, t3);
  k_apply<<<agrid, 256, 0, stream>>>(BUFB, s3, t3, BUFC);   // x0 -> BUFC

  for (int l = 0; l < 5; ++l) {
    const int dil = DIL[l];
    const __bf16* cw0 = cw + (size_t)(l * 2 + 0) * 9 * 4096;
    const __bf16* cw1 = cw + (size_t)(l * 2 + 1) * 9 * 4096;
    const int p0 = (l * 2 + 0) * 64, p1 = (l * 2 + 1) * 64;

    hipMemsetAsync(stats, 0, 512 * sizeof(float), stream);
    k_conv3x3<<<LDIM * 3, 256, 0, stream>>>(BUFC, cw0, rb + p0, dil, BUFA, gsum, gssq);
    k_finalize<<<1, 64, 0, stream>>>(gsum, gssq, rg + p0, rbe + p0, sc, tc);
    k_apply<<<agrid, 256, 0, stream>>>(BUFA, sc, tc, BUFB);

    hipMemsetAsync(stats, 0, 512 * sizeof(float), stream);
    k_conv3x3<<<LDIM * 3, 256, 0, stream>>>(BUFB, cw1, rb + p1, dil, BUFA, gsum, gssq);
    k_finalize<<<1, 64, 0, stream>>>(gsum, gssq, rg + p1, rbe + p1, sc, tc);
    float* dst = (l == 4) ? out : BUFC;                 // in-place residual OK (same idx)
    k_apply_res<<<agrid, 256, 0, stream>>>(BUFA, sc, tc, BUFC, dst);
  }
}
